// Autoencoder_87110526698109
// MI455X (gfx1250) — matrix operand order repaired and checked by reading, not compiled, not run on hardware
//
#include <hip/hip_runtime.h>
#include <hip/hip_bf16.h>

// ---------------- problem sizes -------------------------------------------
#define BATCH   4096
#define NIN     2048
#define NL      16384
#define TOPK    64

typedef __attribute__((ext_vector_type(16))) __bf16 v16bf;
typedef __attribute__((ext_vector_type(8)))  float  v8f;

// =====================================================================
// Kernel 1: xb = bf16(x - pre_bias), row-major [BATCH][NIN]
// =====================================================================
__global__ __launch_bounds__(256) void prep_x_kernel(
    const float* __restrict__ x, const float* __restrict__ pre_bias,
    __bf16* __restrict__ xb)
{
    int idx = (blockIdx.x * 256 + threadIdx.x) * 8;   // 8 contiguous elems
    int k   = idx & (NIN - 1);                        // column (pre_bias idx)
    float4 a  = *(const float4*)(x + idx);
    float4 b  = *(const float4*)(x + idx + 4);
    float4 pa = *(const float4*)(pre_bias + k);
    float4 pb = *(const float4*)(pre_bias + k + 4);
    union { __bf16 h[8]; uint4 u; } o;
    o.h[0] = (__bf16)(a.x - pa.x); o.h[1] = (__bf16)(a.y - pa.y);
    o.h[2] = (__bf16)(a.z - pa.z); o.h[3] = (__bf16)(a.w - pa.w);
    o.h[4] = (__bf16)(b.x - pb.x); o.h[5] = (__bf16)(b.y - pb.y);
    o.h[6] = (__bf16)(b.z - pb.z); o.h[7] = (__bf16)(b.w - pb.w);
    *(uint4*)(xb + idx) = o.u;
}

// =====================================================================
// Kernel 2: ebT[n][k] = bf16(encoder[k][n])  (LDS-tiled transpose)
// =====================================================================
__global__ __launch_bounds__(256) void prep_enc_kernel(
    const float* __restrict__ enc, __bf16* __restrict__ ebT)
{
    __shared__ float tile[32][33];
    int n0 = blockIdx.x * 32;
    int k0 = blockIdx.y * 32;
    int tx = threadIdx.x, ty = threadIdx.y;   // (32, 8)
#pragma unroll
    for (int j = 0; j < 4; ++j) {
        int kl = ty + j * 8;
        tile[kl][tx] = enc[(size_t)(k0 + kl) * NL + n0 + tx];
    }
    __syncthreads();
#pragma unroll
    for (int j = 0; j < 4; ++j) {
        int nl = ty + j * 8;
        ebT[(size_t)(n0 + nl) * NIN + k0 + tx] = (__bf16)tile[tx][nl];
    }
}

// =====================================================================
// Kernel 3: z = xb @ ebT^T + latent_bias  via v_wmma_f32_16x16x32_bf16
//   Block tile 128x128, BK=32, 8 waves (wave32), wave tile 64x32 = 4x2 WMMA.
//   Global->LDS staging via GLOBAL_LOAD_ASYNC_TO_LDS_B128 (ASYNCcnt).
// =====================================================================
__device__ __forceinline__ unsigned lds_off(const void* p) {
    // LDS aperture: generic addr[31:0] = allocation-relative LDS byte offset
    return (unsigned)(size_t)p;
}
__device__ __forceinline__ void async_ld_b128(unsigned lds, const void* g) {
    asm volatile("global_load_async_to_lds_b128 %0, %1, off"
                 :: "v"(lds), "v"(g) : "memory");
}
__device__ __forceinline__ void wait_async0() {
    asm volatile("s_wait_asynccnt 0" ::: "memory");
}

// A fragment 16x32 bf16: lanes 0-15 -> M=lane; VGPR p holds K pair per ISA 7.12.2
__device__ __forceinline__ v16bf frag_a(const __bf16* base, int tm, int mrow, int half) {
    union { v16bf v; unsigned u[8]; } f;
    const __bf16* row = base + (tm + mrow) * 32;
#pragma unroll
    for (int p = 0; p < 8; ++p) {
        int k = (p < 4) ? (2 * p + 8 * half) : (16 + 2 * (p - 4) + 8 * half);
        f.u[p] = *(const unsigned*)(row + k);
    }
    return f.v;
}
// B fragment 32x16 bf16, B stored N-major in LDS (K contiguous per N-row)
__device__ __forceinline__ v16bf frag_b(const __bf16* base, int tn, int nrow, int half) {
    union { v16bf v; unsigned u[8]; } f;
    const __bf16* row = base + (tn + nrow) * 32;
#pragma unroll
    for (int p = 0; p < 8; ++p) {
        int k = (p < 4) ? (2 * p + 8 * half) : (16 + 2 * (p - 4) + 8 * half);
        f.u[p] = *(const unsigned*)(row + k);
    }
    return f.v;
}

__global__ __launch_bounds__(256) void enc_gemm_kernel(
    const __bf16* __restrict__ xb, const __bf16* __restrict__ ebT,
    const float* __restrict__ latent_bias, float* __restrict__ z)
{
    __shared__ __bf16 As[2][128 * 32];   // [m][k]  8 KB each buf
    __shared__ __bf16 Bs[2][128 * 32];   // [n][k]  8 KB each buf

    const int t    = threadIdx.x;
    const int lane = t & 31;
    const int wave = t >> 5;             // 0..7
    const int wm   = wave & 1;           // 2 waves along M
    const int wn   = wave >> 1;          // 4 waves along N
    const int mrow = lane & 15;
    const int half = lane >> 4;
    const int bm0  = blockIdx.y * 128;
    const int bn0  = blockIdx.x * 128;

    // per-thread 16B chunks: c0 = t, c1 = t+256 of 512 chunks per 8KB tile
    const int c0 = t, c1 = t + 256;
    const int r0 = c0 >> 2, o0 = (c0 & 3) << 3;   // row 0..127, elem offset
    const int r1 = c1 >> 2, o1 = (c1 & 3) << 3;

    const __bf16* pA0 = xb  + (size_t)(bm0 + r0) * NIN + o0;
    const __bf16* pA1 = xb  + (size_t)(bm0 + r1) * NIN + o1;
    const __bf16* pB0 = ebT + (size_t)(bn0 + r0) * NIN + o0;
    const __bf16* pB1 = ebT + (size_t)(bn0 + r1) * NIN + o1;

    unsigned lA0[2], lA1[2], lB0[2], lB1[2];
#pragma unroll
    for (int b = 0; b < 2; ++b) {
        lA0[b] = lds_off(&As[b][r0 * 32 + o0]);
        lA1[b] = lds_off(&As[b][r1 * 32 + o1]);
        lB0[b] = lds_off(&Bs[b][r0 * 32 + o0]);
        lB1[b] = lds_off(&Bs[b][r1 * 32 + o1]);
    }

    v8f acc[4][2];
    const v8f vzero = {0, 0, 0, 0, 0, 0, 0, 0};
#pragma unroll
    for (int i = 0; i < 4; ++i)
#pragma unroll
        for (int j = 0; j < 2; ++j) acc[i][j] = vzero;

    // prolog: stage K-chunk 0 into buffer 0 (async, no VGPR data movement)
    async_ld_b128(lA0[0], pA0); async_ld_b128(lA1[0], pA1);
    async_ld_b128(lB0[0], pB0); async_ld_b128(lB1[0], pB1);
    pA0 += 32; pA1 += 32; pB0 += 32; pB1 += 32;

    const int nIter = NIN / 32;          // 64
    for (int it = 0; it < nIter; ++it) {
        // current buffer's 4 async chunks are the only outstanding ops;
        // async loads complete in order -> asynccnt==0 means they landed.
        wait_async0();
        __syncthreads();                 // all waves' chunks visible

        if (it + 1 < nIter) {            // safe: last readers of buf^1 passed barrier
            const int nb = (it + 1) & 1;
            async_ld_b128(lA0[nb], pA0); async_ld_b128(lA1[nb], pA1);
            async_ld_b128(lB0[nb], pB0); async_ld_b128(lB1[nb], pB1);
            pA0 += 32; pA1 += 32; pB0 += 32; pB1 += 32;
        }

        const int buf = it & 1;
        v16bf af[4], bfr[2];
#pragma unroll
        for (int i = 0; i < 4; ++i) af[i]  = frag_a(As[buf], wm * 64 + i * 16, mrow, half);
#pragma unroll
        for (int j = 0; j < 2; ++j) bfr[j] = frag_b(Bs[buf], wn * 32 + j * 16, mrow, half);
#pragma unroll
        for (int i = 0; i < 4; ++i)
#pragma unroll
            for (int j = 0; j < 2; ++j)
                acc[i][j] = __builtin_amdgcn_wmma_f32_16x16x32_bf16(
                    false, af[i], false, bfr[j], (short)0, acc[i][j], false, false);
    }

    // epilogue: C/D layout -> VGPR r holds M = r + 8*half, N = lane&15
#pragma unroll
    for (int j = 0; j < 2; ++j) {
        const int n  = bn0 + wn * 32 + j * 16 + mrow;
        const float lb = latent_bias[n];
#pragma unroll
        for (int i = 0; i < 4; ++i) {
            const int mbase = bm0 + wm * 64 + i * 16 + half * 8;
#pragma unroll
            for (int r = 0; r < 8; ++r)
                z[(size_t)(mbase + r) * NL + n] = acc[i][j][r] + lb;
        }
    }
}

// =====================================================================
// Kernel 4: per-row top-64 radix select (two 8-bit rounds), ReLU, compact
// =====================================================================
__global__ __launch_bounds__(256) void topk_kernel(
    const float* __restrict__ z, float* __restrict__ vals, int* __restrict__ idxs)
{
    const int row = blockIdx.x;
    const int t   = threadIdx.x;
    const float* zr = z + (size_t)row * NL;

    __shared__ unsigned hist[256];
    __shared__ unsigned suf[256];
    __shared__ unsigned sh_b0, sh_have, sh_b1, sh_D, sh_slotD, sh_slotT;

    unsigned keys[64];
#pragma unroll
    for (int i = 0; i < 64; ++i) {
        unsigned u = __float_as_uint(zr[i * 256 + t]);
        keys[i] = (u & 0x80000000u) ? ~u : (u | 0x80000000u); // monotone key
    }

    // round 1: top byte
    hist[t] = 0; __syncthreads();
#pragma unroll
    for (int i = 0; i < 64; ++i) atomicAdd(&hist[keys[i] >> 24], 1u);
    __syncthreads();
    {
        unsigned s = 0;
        for (int j = t + 1; j < 256; ++j) s += hist[j];
        suf[t] = s;
    }
    __syncthreads();
    if (suf[t] < TOPK && suf[t] + hist[t] >= TOPK) { sh_b0 = t; sh_have = TOPK - suf[t]; }
    __syncthreads();
    const unsigned b0 = sh_b0, have = sh_have;

    // round 2: second byte among keys with top byte == b0
    hist[t] = 0; __syncthreads();
#pragma unroll
    for (int i = 0; i < 64; ++i)
        if ((keys[i] >> 24) == b0) atomicAdd(&hist[(keys[i] >> 16) & 255], 1u);
    __syncthreads();
    {
        unsigned s = 0;
        for (int j = t + 1; j < 256; ++j) s += hist[j];
        suf[t] = s;
    }
    __syncthreads();
    if (suf[t] < have && suf[t] + hist[t] >= have) { sh_b1 = t; sh_D = (TOPK - have) + suf[t]; }
    if (t == 0) { sh_slotD = 0; sh_slotT = 0; }
    __syncthreads();
    const unsigned T16 = (b0 << 8) | sh_b1;
    const unsigned D   = sh_D;

    // compaction: >T16 definite, ==T16 fill remaining slots
#pragma unroll
    for (int i = 0; i < 64; ++i) {
        unsigned k16 = keys[i] >> 16;
        if (k16 < T16) continue;
        unsigned slot;
        if (k16 > T16) {
            slot = atomicAdd(&sh_slotD, 1u);
        } else {
            slot = D + atomicAdd(&sh_slotT, 1u);
            if (slot >= TOPK) continue;
        }
        unsigned key = keys[i];
        unsigned u = (key & 0x80000000u) ? (key ^ 0x80000000u) : ~key;
        float v = __uint_as_float(u);
        vals[row * TOPK + slot] = fmaxf(v, 0.0f);   // ReLU on kept values
        idxs[row * TOPK + slot] = i * 256 + t;
    }
}

// =====================================================================
// Kernel 5: recons[row,:] = pre_bias + sum_j vals[j] * decoder[idx[j],:]
// =====================================================================
__global__ __launch_bounds__(256) void decode_kernel(
    const float* __restrict__ dec, const float* __restrict__ pre_bias,
    const float* __restrict__ vals, const int* __restrict__ idxs,
    float* __restrict__ out)
{
    const int row = blockIdx.x;
    const int t   = threadIdx.x;
    __shared__ float sv[TOPK];
    __shared__ int   si[TOPK];
    if (t < TOPK) { sv[t] = vals[row * TOPK + t]; si[t] = idxs[row * TOPK + t]; }
    __syncthreads();

    const int col = t * 8;
    float4 acc0 = *(const float4*)(pre_bias + col);
    float4 acc1 = *(const float4*)(pre_bias + col + 4);
#pragma unroll 8
    for (int j = 0; j < TOPK; ++j) {
        float v = sv[j];                       // uniform across block
        if (v != 0.0f) {
            const float* drow = dec + (size_t)si[j] * NIN + col;
            float4 d0 = *(const float4*)(drow);
            float4 d1 = *(const float4*)(drow + 4);
            acc0.x += v * d0.x; acc0.y += v * d0.y; acc0.z += v * d0.z; acc0.w += v * d0.w;
            acc1.x += v * d1.x; acc1.y += v * d1.y; acc1.z += v * d1.z; acc1.w += v * d1.w;
        }
    }
    *(float4*)(out + (size_t)row * NIN + col)     = acc0;
    *(float4*)(out + (size_t)row * NIN + col + 4) = acc1;
}

// =====================================================================
// Launcher
// =====================================================================
extern "C" void kernel_launch(void* const* d_in, const int* in_sizes, int n_in,
                              void* d_out, int out_size, void* d_ws, size_t ws_size,
                              hipStream_t stream)
{
    (void)in_sizes; (void)n_in; (void)out_size; (void)ws_size;
    const float* x           = (const float*)d_in[0];
    const float* enc         = (const float*)d_in[1];
    const float* dec         = (const float*)d_in[2];
    const float* pre_bias    = (const float*)d_in[3];
    const float* latent_bias = (const float*)d_in[4];
    float* out = (float*)d_out;

    static const size_t XB_BYTES   = (size_t)BATCH * NIN * 2;   // 16 MiB
    static const size_t EBT_BYTES  = (size_t)NIN * NL * 2;      // 64 MiB
    static const size_t Z_BYTES    = (size_t)BATCH * NL * 4;    // 256 MiB
    static const size_t VALS_BYTES = (size_t)BATCH * TOPK * 4;  // 1 MiB

    char*   ws   = (char*)d_ws;
    __bf16* xb   = (__bf16*)ws;
    __bf16* ebT  = (__bf16*)(ws + XB_BYTES);
    float*  z    = (float*)(ws + XB_BYTES + EBT_BYTES);
    float*  vals = (float*)(ws + XB_BYTES + EBT_BYTES + Z_BYTES);
    int*    idxs = (int*)  (ws + XB_BYTES + EBT_BYTES + Z_BYTES + VALS_BYTES);

    prep_x_kernel  <<<(BATCH * NIN) / (256 * 8), 256, 0, stream>>>(x, pre_bias, xb);
    prep_enc_kernel<<<dim3(NL / 32, NIN / 32), dim3(32, 8), 0, stream>>>(enc, ebT);
    enc_gemm_kernel<<<dim3(NL / 128, BATCH / 128), 256, 0, stream>>>(xb, ebT, latent_bias, z);
    topk_kernel    <<<BATCH, 256, 0, stream>>>(z, vals, idxs);
    decode_kernel  <<<BATCH, 256, 0, stream>>>(dec, pre_bias, vals, idxs, out);
}
